// LstmUnit_8735963480661
// MI455X (gfx1250) — matrix operand order repaired and checked by reading, not compiled, not run on hardware
//
#include <hip/hip_runtime.h>
#include <hip/hip_bf16.h>

// ---------------------------------------------------------------------------
// LSTM on MI455X (gfx1250, wave32, WMMA):
//   phase 1: xW = x @ W + bias   (big parallel bf16 WMMA GEMM, fp32 acc)
//            4 M-tiles per wave -> each B fragment feeds 4 WMMAs
//   phase 2: 512 sequential steps: gates = xW_t + h @ U
//            h_t staged in LDS via async global->LDS copies; U^T read once
//            per step; then pointwise sigmoid/tanh cell update.
// ---------------------------------------------------------------------------

typedef __attribute__((ext_vector_type(16))) __bf16 v16bf;
typedef __attribute__((ext_vector_type(8)))  float  v8f;
typedef __attribute__((ext_vector_type(4)))  float  f4;
typedef __attribute__((ext_vector_type(4)))  int    v4i;

#define BS   64
#define SEQ  512
#define INP  512
#define HS   512
#define GN   2048          // 4*HS gate columns

#if __has_builtin(__builtin_amdgcn_global_load_async_to_lds_b128)
#define HAVE_ASYNC_LDS 1
#endif

// ---- workspace layout (bytes) ---------------------------------------------
#define XBF_OFF   0u                                    // 32768 x 512 bf16  (x, [s][b][k])
#define XBF_SZ    ((size_t)BS * SEQ * INP * 2)
#define WT_OFF    (XBF_OFF + XBF_SZ)                    // 2048 x 512 bf16  (W^T)
#define WT_SZ     ((size_t)GN * INP * 2)
#define UT_OFF    (WT_OFF + WT_SZ)                      // 2048 x 512 bf16  (U^T)
#define UT_SZ     ((size_t)GN * HS * 2)
#define HBF_OFF   (UT_OFF + UT_SZ)                      // 64 x 512 bf16    (h_t)
#define HBF_SZ    ((size_t)BS * HS * 2)
#define C_OFF     (HBF_OFF + HBF_SZ)                    // 64 x 512 f32     (c_t)
#define C_SZ      ((size_t)BS * HS * 4)
#define GATE_OFF  (C_OFF + C_SZ)                        // 64 x 2048 f32    (gates scratch)
#define GATE_SZ   ((size_t)BS * GN * 4)
#define XW_OFF    (GATE_OFF + GATE_SZ)                  // [512][64][2048] f32
#define XW_SZ     ((size_t)SEQ * BS * GN * 4)

// ---------------------------------------------------------------------------
// WMMA fragment loaders (ISA 7.12.2 layouts, wave32)
// A 16x32 bf16: lane<16 -> row (lane), K in [0,8) and [16,24)
//               lane>=16 -> row (lane-16), K in [8,16) and [24,32)
// B 32x16 bf16: lane&15 -> col, same K split as A: (lane>>4)*8 + [0,8) and [16,24)
// ---------------------------------------------------------------------------
__device__ __forceinline__ v16bf load_a_frag(const __hip_bfloat16* __restrict__ A,
                                             int lda, int row0, int k0, int lane) {
    int r  = row0 + (lane & 15);
    int kb = (lane >> 4) * 8;
    const __hip_bfloat16* p = A + (size_t)r * lda + k0 + kb;
    v16bf a;
    ((f4*)&a)[0] = *(const f4*)p;         // K = kb .. kb+7      -> VGPR 0..3
    ((f4*)&a)[1] = *(const f4*)(p + 16);  // K = kb+16 .. kb+23  -> VGPR 4..7
    return a;
}

__device__ __forceinline__ v16bf load_b_frag(const __hip_bfloat16* __restrict__ Bt,
                                             int ldb, int col0, int k0, int lane) {
    int n  = col0 + (lane & 15);
    int kb = (lane >> 4) * 8;
    const __hip_bfloat16* p = Bt + (size_t)n * ldb + k0 + kb;
    v16bf b;
    ((f4*)&b)[0] = *(const f4*)p;         // K = kb .. kb+7      -> VGPR 0..3
    ((f4*)&b)[1] = *(const f4*)(p + 16);  // K = kb+16 .. kb+23  -> VGPR 4..7
    return b;
}

#define WMMA_BF16(a, b, acc) \
    __builtin_amdgcn_wmma_f32_16x16x32_bf16(false, (a), false, (b), (short)0, (acc), false, false)

// async 16-byte global -> LDS copy (falls back to a plain LDS store)
__device__ __forceinline__ void async_copy16(const __hip_bfloat16* g, __hip_bfloat16* l) {
#ifdef HAVE_ASYNC_LDS
    __builtin_amdgcn_global_load_async_to_lds_b128(
        (__attribute__((address_space(1))) v4i*)g,
        (__attribute__((address_space(3))) v4i*)l, 0, 0);
#else
    *(f4*)l = *(const f4*)g;
#endif
}

// ---------------------------------------------------------------------------
// conversion kernels
// ---------------------------------------------------------------------------
__global__ void lstm_cvt_x(const float* __restrict__ x, __hip_bfloat16* __restrict__ xbf) {
    int idx = blockIdx.x * blockDim.x + threadIdx.x;   // b*SEQ*INP + s*INP + k
    int k = idx & (INP - 1);
    int s = (idx >> 9) & (SEQ - 1);
    int b = idx >> 18;
    xbf[(size_t)(s * BS + b) * INP + k] = __float2bfloat16(x[idx]);   // [s][b][k]
}

__global__ void lstm_cvt_t(const float* __restrict__ Min, __hip_bfloat16* __restrict__ Mt) {
    int idx = blockIdx.x * blockDim.x + threadIdx.x;   // k*GN + n over [512][2048]
    int n = idx & (GN - 1);
    int k = idx >> 11;
    Mt[(size_t)n * INP + k] = __float2bfloat16(Min[idx]);             // [n][k]
}

__global__ void lstm_init_state(float* __restrict__ c, __hip_bfloat16* __restrict__ hbf) {
    int idx = blockIdx.x * blockDim.x + threadIdx.x;   // 0 .. BS*HS-1
    c[idx]   = 0.0f;
    hbf[idx] = __float2bfloat16(0.0f);
}

// ---------------------------------------------------------------------------
// phase 1: xW[m][n] = sum_k xbf[m][k] * Wt[n][k] + bias[n]
// each wave computes a 64x16 slab (4 M-tiles, 1 N-tile): B reused 4x
// ---------------------------------------------------------------------------
__global__ void lstm_gemm_xw(const __hip_bfloat16* __restrict__ xbf,
                             const __hip_bfloat16* __restrict__ Wt,
                             const float* __restrict__ bias,
                             float* __restrict__ xW) {
    const int NT = GN / 16;                                  // 128 column tiles
    int wave = (blockIdx.x * blockDim.x + threadIdx.x) >> 5; // wave-uniform
    int lane = threadIdx.x & 31;
    int mg = wave / NT;                                      // 0..511 (64-row slab)
    int nt = wave - mg * NT;
    int row0 = mg * 64, col0 = nt * 16;
    int cn   = col0 + (lane & 15);
    int rsub = (lane >> 4) * 8;

    v8f acc0, acc1, acc2, acc3;
    float binit = bias[cn];
#pragma unroll
    for (int v = 0; v < 8; ++v) { acc0[v] = binit; acc1[v] = binit; acc2[v] = binit; acc3[v] = binit; }

#pragma unroll 2
    for (int k0 = 0; k0 < INP; k0 += 32) {
        v16bf b  = load_b_frag(Wt,  INP, col0, k0, lane);
        v16bf a0 = load_a_frag(xbf, INP, row0,      k0, lane);
        v16bf a1 = load_a_frag(xbf, INP, row0 + 16, k0, lane);
        v16bf a2 = load_a_frag(xbf, INP, row0 + 32, k0, lane);
        v16bf a3 = load_a_frag(xbf, INP, row0 + 48, k0, lane);
        acc0 = WMMA_BF16(a0, b, acc0);
        acc1 = WMMA_BF16(a1, b, acc1);
        acc2 = WMMA_BF16(a2, b, acc2);
        acc3 = WMMA_BF16(a3, b, acc3);
    }

    int rb = row0 + rsub;                                    // C/D layout
#pragma unroll
    for (int v = 0; v < 8; ++v) {
        xW[(size_t)(rb + v)      * GN + cn] = acc0[v];
        xW[(size_t)(rb + 16 + v) * GN + cn] = acc1[v];
        xW[(size_t)(rb + 32 + v) * GN + cn] = acc2[v];
        xW[(size_t)(rb + 48 + v) * GN + cn] = acc3[v];
    }
}

// ---------------------------------------------------------------------------
// phase 2a: gates[m][n] = xW_t[m][n] + sum_k h[m][k] * Ut[n][k]
// h_t (64 KB) async-staged into LDS once per block; each wave computes a
// 64x16 slab so every U^T element is read exactly once per timestep.
// 128 N-tiles / 8 waves -> 16 blocks of 256 threads.
// ---------------------------------------------------------------------------
__global__ void lstm_gemm_step(const __hip_bfloat16* __restrict__ hbf,
                               const __hip_bfloat16* __restrict__ Ut,
                               const float* __restrict__ xWt,
                               float* __restrict__ gates) {
    __shared__ __hip_bfloat16 sh[BS * HS];                   // 64 KB staged h_t

    int tid = threadIdx.x;
#pragma unroll
    for (int i = 0; i < 16; ++i) {                           // 256B per thread
        int off = (i * 256 + tid) * 8;                       // 16B chunks
        async_copy16(hbf + off, &sh[off]);
    }
#ifdef HAVE_ASYNC_LDS
    asm volatile("s_wait_asynccnt 0x0" ::: "memory");
#endif
    __syncthreads();

    int wave = tid >> 5;                                     // 0..7 (uniform)
    int lane = tid & 31;
    int nt   = blockIdx.x * 8 + wave;                        // 0..127
    int col0 = nt * 16;
    int cn   = col0 + (lane & 15);
    int rsub = (lane >> 4) * 8;

    v8f acc0, acc1, acc2, acc3;
#pragma unroll
    for (int v = 0; v < 8; ++v) {
        acc0[v] = xWt[(size_t)(rsub + v)      * GN + cn];
        acc1[v] = xWt[(size_t)(rsub + 16 + v) * GN + cn];
        acc2[v] = xWt[(size_t)(rsub + 32 + v) * GN + cn];
        acc3[v] = xWt[(size_t)(rsub + 48 + v) * GN + cn];
    }

#pragma unroll 2
    for (int k0 = 0; k0 < HS; k0 += 32) {
        v16bf b  = load_b_frag(Ut, HS, col0, k0, lane);      // global (L2-resident)
        v16bf a0 = load_a_frag(sh, HS, 0,  k0, lane);        // LDS ds_load
        v16bf a1 = load_a_frag(sh, HS, 16, k0, lane);
        v16bf a2 = load_a_frag(sh, HS, 32, k0, lane);
        v16bf a3 = load_a_frag(sh, HS, 48, k0, lane);
        acc0 = WMMA_BF16(a0, b, acc0);
        acc1 = WMMA_BF16(a1, b, acc1);
        acc2 = WMMA_BF16(a2, b, acc2);
        acc3 = WMMA_BF16(a3, b, acc3);
    }

#pragma unroll
    for (int v = 0; v < 8; ++v) {
        gates[(size_t)(rsub + v)      * GN + cn] = acc0[v];
        gates[(size_t)(rsub + 16 + v) * GN + cn] = acc1[v];
        gates[(size_t)(rsub + 32 + v) * GN + cn] = acc2[v];
        gates[(size_t)(rsub + 48 + v) * GN + cn] = acc3[v];
    }
}

// ---------------------------------------------------------------------------
// phase 2b: pointwise cell update. one thread per (b, j), 32768 threads.
// ---------------------------------------------------------------------------
__global__ void lstm_pointwise(const float* __restrict__ gates,
                               float* __restrict__ c,
                               float* __restrict__ hseq,
                               __hip_bfloat16* __restrict__ hbf,
                               int t) {
    int idx = blockIdx.x * blockDim.x + threadIdx.x;   // b*HS + j
    int b = idx >> 9;
    int j = idx & (HS - 1);
    const float* g = gates + (size_t)b * GN;

    float ig = 1.0f / (1.0f + __expf(-g[j]));
    float fg = 1.0f / (1.0f + __expf(-(g[HS + j] + 1.0f)));   // forget bias +1
    float gg = tanhf(g[2 * HS + j]);
    float og = 1.0f / (1.0f + __expf(-g[3 * HS + j]));

    float cn = fg * c[idx] + ig * gg;
    float h  = og * tanhf(cn);

    c[idx] = cn;
    hseq[((size_t)b * SEQ + t) * HS + j] = h;                 // [bs][seq][HS]
    hbf[idx] = __float2bfloat16(h);                           // next step's A matrix
}

// final tuple members: h_T = hidden[:, SEQ-1, :], c_T
__global__ void lstm_final(const float* __restrict__ hseq,
                           const float* __restrict__ c,
                           float* __restrict__ tail) {
    int idx = blockIdx.x * blockDim.x + threadIdx.x;   // 0 .. BS*HS-1
    int b = idx >> 9;
    int j = idx & (HS - 1);
    tail[idx]           = hseq[((size_t)b * SEQ + (SEQ - 1)) * HS + j];
    tail[BS * HS + idx] = c[idx];
}

// ---------------------------------------------------------------------------
extern "C" void kernel_launch(void* const* d_in, const int* in_sizes, int n_in,
                              void* d_out, int out_size, void* d_ws, size_t ws_size,
                              hipStream_t stream) {
    const float* x    = (const float*)d_in[0];   // [64][512][512]
    const float* W    = (const float*)d_in[1];   // [512][2048]
    const float* U    = (const float*)d_in[2];   // [512][2048]
    const float* bias = (const float*)d_in[3];   // [2048]
    float* out = (float*)d_out;                  // hidden_seq | h_T | c_T

    char* ws = (char*)d_ws;
    __hip_bfloat16* xbf = (__hip_bfloat16*)(ws + XBF_OFF);
    __hip_bfloat16* Wt  = (__hip_bfloat16*)(ws + WT_OFF);
    __hip_bfloat16* Ut  = (__hip_bfloat16*)(ws + UT_OFF);
    __hip_bfloat16* hbf = (__hip_bfloat16*)(ws + HBF_OFF);
    float* c     = (float*)(ws + C_OFF);
    float* gates = (float*)(ws + GATE_OFF);
    float* xW    = (float*)(ws + XW_OFF);

    // conversions + state init
    lstm_cvt_x<<<(BS * SEQ * INP) / 256, 256, 0, stream>>>(x, xbf);
    lstm_cvt_t<<<(INP * GN) / 256, 256, 0, stream>>>(W, Wt);
    lstm_cvt_t<<<(HS * GN) / 256, 256, 0, stream>>>(U, Ut);
    lstm_init_state<<<(BS * HS) / 256, 256, 0, stream>>>(c, hbf);

    // phase 1: (32768/64 M-slabs) * (2048/16 N-tiles) waves, 8 waves/block
    lstm_gemm_xw<<<((BS * SEQ / 64) * (GN / 16)) / 8, 256, 0, stream>>>(xbf, Wt, bias, xW);

    // phase 2: sequential scan
    for (int t = 0; t < SEQ; ++t) {
        lstm_gemm_step<<<(GN / 16) / 8, 256, 0, stream>>>(
            hbf, Ut, xW + (size_t)t * BS * GN, gates);
        lstm_pointwise<<<(BS * HS) / 256, 256, 0, stream>>>(gates, c, out, hbf, t);
    }

    lstm_final<<<(BS * HS) / 256, 256, 0, stream>>>(out, c, out + (size_t)BS * SEQ * HS);
}
